// RGCNLayer_20641612825462
// MI455X (gfx1250) — matrix operand order repaired and checked by reading, not compiled, not run on hardware
//
#include <hip/hip_runtime.h>

#define NN   10000   // nodes
#define NE   320000  // edges
#define DIN  256
#define DOUT 256
#define NR   16

typedef __attribute__((ext_vector_type(16))) _Float16 v16h;
typedef __attribute__((ext_vector_type(8)))  float    v8f;

// ---------------------------------------------------------------------------
// zero the per-(relation,dst) degree buffer
// ---------------------------------------------------------------------------
__global__ __launch_bounds__(256) void rgcn_zero_deg(float* __restrict__ deg) {
    int i = blockIdx.x * blockDim.x + threadIdx.x;
    if (i < NR * NN) deg[i] = 0.0f;
}

// ---------------------------------------------------------------------------
// deg[r, dst] += 1 per edge
// ---------------------------------------------------------------------------
__global__ __launch_bounds__(256) void rgcn_deg_count(const int* __restrict__ dst,
                                                      const int* __restrict__ et,
                                                      float* __restrict__ deg) {
    int e = blockIdx.x * blockDim.x + threadIdx.x;
    if (e < NE) {
        atomicAdd(deg + et[e] * NN + dst[e], 1.0f);
    }
}

// ---------------------------------------------------------------------------
// helper: build 16x32 f16 A-fragment from a row-major fp32 row
// lanes 0-15 (half=0): K = k0+{0..15} ; lanes 16-31: K = k0+{16..31}
// kbase = k0 + half*16 (64B aligned -> float4 loads legal)
// ---------------------------------------------------------------------------
__device__ __forceinline__ v16h load_a_frag(const float* __restrict__ xrow, int kbase) {
    float4 f0 = *reinterpret_cast<const float4*>(xrow + kbase);
    float4 f1 = *reinterpret_cast<const float4*>(xrow + kbase + 4);
    float4 f2 = *reinterpret_cast<const float4*>(xrow + kbase + 8);
    float4 f3 = *reinterpret_cast<const float4*>(xrow + kbase + 12);
    v16h a;
    a[0] = (_Float16)f0.x;  a[1] = (_Float16)f0.y;  a[2]  = (_Float16)f0.z;  a[3]  = (_Float16)f0.w;
    a[4] = (_Float16)f1.x;  a[5] = (_Float16)f1.y;  a[6]  = (_Float16)f1.z;  a[7]  = (_Float16)f1.w;
    a[8] = (_Float16)f2.x;  a[9] = (_Float16)f2.y;  a[10] = (_Float16)f2.z;  a[11] = (_Float16)f2.w;
    a[12] = (_Float16)f3.x; a[13] = (_Float16)f3.y; a[14] = (_Float16)f3.z;  a[15] = (_Float16)f3.w;
    return a;
}

__device__ __forceinline__ v8f wmma_f16(v16h a, v16h b, v8f c) {
    return __builtin_amdgcn_wmma_f32_16x16x32_f16(false, a, false, b, (short)0, c,
                                                  false, false);
}

// ---------------------------------------------------------------------------
// h_all[r, n, o] = sum_d x[n,d] * W_r[r,d,o]   (stored fp16, fp32 accumulate)
// One wave per 16(M) x 32(N) output strip, grid = (625, 8, 16).
// W tile is staged through LDS: coalesced float4 global loads -> f16 LDS tile
// -> per-lane fragment gathers. A-fragment is reused by both accumulators.
// ---------------------------------------------------------------------------
__global__ __launch_bounds__(32) void rgcn_rel_gemm(const float* __restrict__ x,
                                                    const float* __restrict__ W_r,
                                                    _Float16* __restrict__ h_all) {
    __shared__ _Float16 bt[32][32];   // [k - k0][n - n0], 2 KB

    const int mt = blockIdx.x, ng = blockIdx.y, r = blockIdx.z;
    const int lane = threadIdx.x;
    const int half = lane >> 4;
    const int lr   = lane & 15;
    const int n0   = ng * 32;

    const float* xrow = x + (size_t)(mt * 16 + lr) * DIN;
    const float* Wrel = W_r + (size_t)r * DIN * DOUT;

    // staging coordinates: 8 lanes x 4 cols span one 32-col row chunk
    const int cg = (lane & 7) * 4;    // column offset within tile
    const int kr = lane >> 3;         // row offset 0..3 within a 4-row group

    v8f c0 = {}, c1 = {};
    for (int k0 = 0; k0 < DIN; k0 += 32) {
        // ---- stage B tile (32k x 32n) into LDS, converting f32 -> f16 ----
#pragma unroll
        for (int i = 0; i < 8; ++i) {
            const int kk = i * 4 + kr;
            float4 w = *reinterpret_cast<const float4*>(
                Wrel + (size_t)(k0 + kk) * DOUT + n0 + cg);
            _Float16* d = &bt[kk][cg];
            d[0] = (_Float16)w.x; d[1] = (_Float16)w.y;
            d[2] = (_Float16)w.z; d[3] = (_Float16)w.w;
        }
        // prefetch next K-step's W rows into cache while we compute
        if (k0 + 32 < DIN) {
            __builtin_prefetch(Wrel + (size_t)(k0 + 32 + kr) * DOUT + n0 + cg, 0, 1);
        }
        __syncthreads();

        // ---- fragments ----
        v16h a = load_a_frag(xrow, k0 + half * 16);
        v16h b0, b1;
#pragma unroll
        for (int j = 0; j < 16; ++j) {
            b0[j] = bt[half * 16 + j][lr];        // columns n0 + 0..15
            b1[j] = bt[half * 16 + j][16 + lr];   // columns n0 + 16..31
        }
        __syncthreads();

        c0 = wmma_f16(a, b0, c0);
        c1 = wmma_f16(a, b1, c1);
    }

    // C/D layout: vgpr v -> row (half*8 + v), lane lr -> col
#pragma unroll
    for (int v = 0; v < 8; ++v) {
        const int m = mt * 16 + half * 8 + v;
        const size_t base = ((size_t)r * NN + m) * DOUT;
        h_all[base + n0 + lr]      = (_Float16)c0[v];
        h_all[base + n0 + 16 + lr] = (_Float16)c1[v];
    }
}

// ---------------------------------------------------------------------------
// out[n, o] = sum_d x[n,d] * W0_w[o,d] + b[o]    (initializes out; no zeroing)
// W0_w is [DOUT, DIN] so the B-fragment (column o, consecutive d) is contiguous.
// ---------------------------------------------------------------------------
__global__ __launch_bounds__(32) void rgcn_selfloop(const float* __restrict__ x,
                                                    const float* __restrict__ W0w,
                                                    const float* __restrict__ W0b,
                                                    float* __restrict__ out) {
    const int mt = blockIdx.x, nt = blockIdx.y;
    const int lane = threadIdx.x;
    const int half = lane >> 4;
    const int lr   = lane & 15;

    const float* xrow = x + (size_t)(mt * 16 + lr) * DIN;
    const int    col  = nt * 16 + lr;

    v8f c = {};
    for (int k0 = 0; k0 < DIN; k0 += 32) {
        v16h a = load_a_frag(xrow, k0 + half * 16);
        // b[j] = W0w[col, k0 + half*16 + j] -- 16 contiguous floats, 64B aligned
        const float* bp = W0w + (size_t)col * DIN + k0 + half * 16;
        float4 g0 = *reinterpret_cast<const float4*>(bp);
        float4 g1 = *reinterpret_cast<const float4*>(bp + 4);
        float4 g2 = *reinterpret_cast<const float4*>(bp + 8);
        float4 g3 = *reinterpret_cast<const float4*>(bp + 12);
        v16h b;
        b[0] = (_Float16)g0.x;  b[1] = (_Float16)g0.y;  b[2]  = (_Float16)g0.z;  b[3]  = (_Float16)g0.w;
        b[4] = (_Float16)g1.x;  b[5] = (_Float16)g1.y;  b[6]  = (_Float16)g1.z;  b[7]  = (_Float16)g1.w;
        b[8] = (_Float16)g2.x;  b[9] = (_Float16)g2.y;  b[10] = (_Float16)g2.z;  b[11] = (_Float16)g2.w;
        b[12] = (_Float16)g3.x; b[13] = (_Float16)g3.y; b[14] = (_Float16)g3.z;  b[15] = (_Float16)g3.w;
        c = wmma_f16(a, b, c);
    }
    float bias = W0b[nt * 16 + lr];
#pragma unroll
    for (int v = 0; v < 8; ++v) {
        int m = mt * 16 + half * 8 + v;
        out[(size_t)m * DOUT + nt * 16 + lr] = c[v] + bias;
    }
}

// ---------------------------------------------------------------------------
// per-edge: out[dst, :] += h_all[et, src, :] / max(deg[et,dst],1)
// one block of 256 threads per edge; h_all (82MB fp16) is L2-resident.
// ---------------------------------------------------------------------------
__global__ __launch_bounds__(256) void rgcn_edge_scatter(const _Float16* __restrict__ h_all,
                                                         const float* __restrict__ deg,
                                                         const int* __restrict__ src,
                                                         const int* __restrict__ dst,
                                                         const int* __restrict__ et,
                                                         float* __restrict__ out) {
    const int e = blockIdx.x;
    const int t = threadIdx.x;
    const int s = src[e];
    const int d = dst[e];
    const int r = et[e];
    const float inv = 1.0f / fmaxf(deg[r * NN + d], 1.0f);
    const float m = (float)h_all[((size_t)r * NN + s) * DOUT + t] * inv;
    atomicAdd(out + (size_t)d * DOUT + t, m);
}

// ---------------------------------------------------------------------------
extern "C" void kernel_launch(void* const* d_in, const int* in_sizes, int n_in,
                              void* d_out, int out_size, void* d_ws, size_t ws_size,
                              hipStream_t stream) {
    const float* x    = (const float*)d_in[0];   // [NN, DIN]
    const float* W_r  = (const float*)d_in[1];   // [NR, DIN, DOUT]
    const float* W0w  = (const float*)d_in[2];   // [DOUT, DIN]
    const float* W0b  = (const float*)d_in[3];   // [DOUT]
    const int*   eidx = (const int*)d_in[4];     // [2, NE] (src row, dst row)
    const int*   et   = (const int*)d_in[5];     // [NE]
    float*       out  = (float*)d_out;           // [NN, DOUT]

    const int* src = eidx;
    const int* dst = eidx + NE;

    float*    deg   = (float*)d_ws;                                               // NR*NN f32
    _Float16* h_all = (_Float16*)((char*)d_ws + (size_t)NR * NN * sizeof(float)); // NR*NN*DOUT f16

    // 1. zero degree buffer, count per-(relation,dst) in-degree
    rgcn_zero_deg<<<(NR * NN + 255) / 256, 256, 0, stream>>>(deg);
    rgcn_deg_count<<<(NE + 255) / 256, 256, 0, stream>>>(dst, et, deg);

    // 2. self-loop GEMM initializes out = x @ W0^T + b  (WMMA)
    rgcn_selfloop<<<dim3(NN / 16, DOUT / 16), 32, 0, stream>>>(x, W0w, W0b, out);

    // 3. grouped GEMM h_all[r] = x @ W_r[r]  (WMMA, LDS-staged B, fp16 output)
    rgcn_rel_gemm<<<dim3(NN / 16, DOUT / 32, NR), 32, 0, stream>>>(x, W_r, h_all);

    // 4. normalized gather/scatter over edges
    rgcn_edge_scatter<<<NE, 256, 0, stream>>>(h_all, deg, src, dst, et, out);
}
